// AutoregressiveLSTM_83648783057126
// MI455X (gfx1250) — compile-verified, matrix operand order checked by reading, not run on hardware
//
#include <hip/hip_runtime.h>
#include <stdint.h>
#include <stddef.h>

// ---------------------------------------------------------------------------
// Autoregressive 2-layer zoneout LSTM for MI455X (gfx1250).
// Strategy:
//   * pre-pass: weights fp32 -> bf16 (L2-resident, ~30MB), inputs concat+shift
//     -> bf16 [T][B][608]
//   * persistent kernel, 64 WGs x 256 thr = 512 wave32 waves, grid barrier
//     4x per timestep; all matmuls via v_wmma_f32_16x16x32_bf16
// ---------------------------------------------------------------------------

typedef __bf16 bf16_t;
typedef bf16_t v16bf __attribute__((ext_vector_type(16)));
typedef float  v8f   __attribute__((ext_vector_type(8)));

#define NB   32      // batch
#define ND   80      // mel dim
#define NT   1000    // time steps
#define NC   512     // cond dim
#define NH   1024    // hidden
#define NG   4096    // 4*H (gates)
#define K0P  608     // (D + C) padded to multiple of 32
#define NWG  64      // persistent workgroups
#define NTHR 256     // threads per workgroup (8 waves)

// ---- workspace layout (bytes) ---------------------------------------------
constexpr size_t OFF_WIH0  = 0;
constexpr size_t SZ_WIH0   = (size_t)NG * K0P * 2;
constexpr size_t OFF_WHH0  = OFF_WIH0 + SZ_WIH0;
constexpr size_t SZ_WHH    = (size_t)NG * NH * 2;
constexpr size_t OFF_WIH1  = OFF_WHH0 + SZ_WHH;
constexpr size_t OFF_WHH1  = OFF_WIH1 + SZ_WHH;
constexpr size_t OFF_PROJW = OFF_WHH1 + SZ_WHH;
constexpr size_t SZ_PROJW  = (size_t)ND * NH * 2;
constexpr size_t OFF_XIN   = OFF_PROJW + SZ_PROJW;
constexpr size_t SZ_XIN    = (size_t)NT * NB * K0P * 2;
constexpr size_t OFF_H0B   = OFF_XIN + SZ_XIN;         // bf16 h0
constexpr size_t SZ_HB     = (size_t)NB * NH * 2;
constexpr size_t OFF_H1B   = OFF_H0B + SZ_HB;          // bf16 h1
constexpr size_t OFF_H0F   = OFF_H1B + SZ_HB;          // fp32 h0
constexpr size_t SZ_HF     = (size_t)NB * NH * 4;
constexpr size_t OFF_C0F   = OFF_H0F + SZ_HF;
constexpr size_t OFF_H1F   = OFF_C0F + SZ_HF;
constexpr size_t OFF_C1F   = OFF_H1F + SZ_HF;
constexpr size_t OFF_BAR   = OFF_C1F + SZ_HF;          // grid barrier (cnt, gen)
constexpr size_t SZ_BAR    = 256;
constexpr size_t OFF_G0    = OFF_BAR + SZ_BAR;         // gate scratch layer 0
constexpr size_t SZ_G      = (size_t)NB * NG * 4;
constexpr size_t OFF_G1    = OFF_G0 + SZ_G;
constexpr size_t ZERO_OFF  = OFF_H0B;
constexpr size_t ZERO_BYTES= OFF_G0 - OFF_H0B;         // state + barrier

// ---- helpers --------------------------------------------------------------
__device__ __forceinline__ bf16_t f2bf(float f) {
  unsigned u = __builtin_bit_cast(unsigned, f);
  unsigned r = u + 0x7FFFu + ((u >> 16) & 1u);       // round-to-nearest-even
  unsigned short hi = (unsigned short)(r >> 16);
  return __builtin_bit_cast(bf16_t, hi);
}

__device__ __forceinline__ v8f wmma_bf16(v16bf a, v16bf b, v8f c) {
  // D = A(16x32 bf16) x B(32x16 bf16) + C(16x16 f32)
  return __builtin_amdgcn_wmma_f32_16x16x32_bf16(
      /*neg_a=*/false, a, /*neg_b=*/false, b,
      /*c_mod=*/(short)0, c, /*reuse_a=*/false, /*reuse_b=*/false);
}

// Load one lane's share of a 16x32 bf16 operand (A or B^T) from a row-major,
// K-contiguous matrix. rowp = this lane's row base pointer; k0 = K offset.
// ISA layout: lanes 0-15 hold K = k0+[0..7] and k0+[16..23];
//             lanes 16-31 hold K = k0+[8..15] and k0+[24..31].
__device__ __forceinline__ v16bf load_frag(const bf16_t* __restrict__ rowp, int k0) {
  const int h = (threadIdx.x >> 4) & 1;
  const uint4* p = (const uint4*)(rowp + k0 + h * 8);
  union { uint4 q[2]; v16bf v; } u;
  u.q[0] = p[0];   // 16 bytes: K = k0 + h*8 .. +7
  u.q[1] = p[2];   // 16 bytes: K = k0 + 16 + h*8 .. +7
  return u.v;
}

// One wave computes one 16x16 tile of gates = A1 @ W1^T + A2 @ W2^T.
// Tasks: wid in [0,512): mt = M-tile (0/1), then 4 gates x 64 column strips.
__device__ __forceinline__ void gemm_gates(
    const bf16_t* __restrict__ A1, int K1, const bf16_t* __restrict__ W1,
    const bf16_t* __restrict__ A2, const bf16_t* __restrict__ W2,
    float* __restrict__ gates, int wid, int lane)
{
  const int mt   = wid & 1;
  const int rest = wid >> 1;                    // 0..255
  const int n0   = (rest >> 6) * NH + (rest & 63) * 16;  // gate*1024 + strip*16
  const int r    = lane & 15;

  v8f acc;
#pragma unroll
  for (int i = 0; i < 8; ++i) acc[i] = 0.0f;

  const bf16_t* a1 = A1 + (size_t)(mt * 16 + r) * K1;
  const bf16_t* w1 = W1 + (size_t)(n0 + r) * K1;
  for (int k0 = 0; k0 < K1; k0 += 32) {
    __builtin_prefetch((const void*)(w1 + k0 + 256), 0, 1);
    acc = wmma_bf16(load_frag(a1, k0), load_frag(w1, k0), acc);
  }
  const bf16_t* a2 = A2 + (size_t)(mt * 16 + r) * NH;
  const bf16_t* w2 = W2 + (size_t)(n0 + r) * NH;
  for (int k0 = 0; k0 < NH; k0 += 32) {
    __builtin_prefetch((const void*)(w2 + k0 + 256), 0, 1);
    acc = wmma_bf16(load_frag(a2, k0), load_frag(w2, k0), acc);
  }
  // C layout: lane -> column n0+(lane&15); VGPR rr -> row mt*16 + 8*(lane>>4) + rr
  float* gp = gates + (size_t)(mt * 16 + ((lane >> 4) & 1) * 8) * NG + n0 + r;
#pragma unroll
  for (int rr = 0; rr < 8; ++rr) gp[(size_t)rr * NG] = acc[rr];
}

// Zoneout LSTM cell pointwise update; 16384 grid threads x 2 elements.
__device__ __forceinline__ void cell_pointwise(
    const float* __restrict__ g,
    const float* __restrict__ bih, const float* __restrict__ bhh,
    float* __restrict__ hf, float* __restrict__ cf, bf16_t* __restrict__ hb,
    int gtid)
{
  for (int e = gtid; e < NB * NH; e += NWG * NTHR) {
    const int b = e >> 10, j = e & (NH - 1);
    const float* gr = g + (size_t)b * NG;
    float gi = gr[j]          + bih[j]          + bhh[j];
    float gf = gr[j + NH]     + bih[j + NH]     + bhh[j + NH];
    float gg = gr[j + 2 * NH] + bih[j + 2 * NH] + bhh[j + 2 * NH];
    float go = gr[j + 3 * NH] + bih[j + 3 * NH] + bhh[j + 3 * NH];
    float co = cf[e], ho = hf[e];
    float si = 1.0f / (1.0f + __expf(-gi));
    float sf = 1.0f / (1.0f + __expf(-gf));
    float so = 1.0f / (1.0f + __expf(-go));
    float cn = sf * co + si * tanhf(gg);
    float hn = so * tanhf(cn);
    float h  = 0.1f * ho + 0.9f * hn;   // eval-mode zoneout
    float c  = 0.1f * co + 0.9f * cn;
    hf[e] = h; cf[e] = c; hb[e] = f2bf(h);
  }
}

// Grid-wide sense barrier over device memory (persistent, all WGs resident).
__device__ __forceinline__ void grid_sync(unsigned* cnt, unsigned* gen) {
  __threadfence();
  __syncthreads();
  if (threadIdx.x == 0) {
    unsigned g = __hip_atomic_load(gen, __ATOMIC_ACQUIRE, __HIP_MEMORY_SCOPE_AGENT);
    unsigned a = __hip_atomic_fetch_add(cnt, 1u, __ATOMIC_ACQ_REL, __HIP_MEMORY_SCOPE_AGENT);
    if (a == NWG - 1) {
      __hip_atomic_store(cnt, 0u, __ATOMIC_RELAXED, __HIP_MEMORY_SCOPE_AGENT);
      __hip_atomic_fetch_add(gen, 1u, __ATOMIC_ACQ_REL, __HIP_MEMORY_SCOPE_AGENT);
    } else {
      while (__hip_atomic_load(gen, __ATOMIC_ACQUIRE, __HIP_MEMORY_SCOPE_AGENT) == g)
        __builtin_amdgcn_s_sleep(1);
    }
  }
  __syncthreads();
  __threadfence();
}

// ---- pre-pass kernels ------------------------------------------------------
__global__ void zero_ws(float* p, size_t n) {
  size_t i = (size_t)blockIdx.x * blockDim.x + threadIdx.x;
  size_t s = (size_t)gridDim.x * blockDim.x;
  for (; i < n; i += s) p[i] = 0.0f;
}

__global__ void convert_pad(const float* __restrict__ src, bf16_t* __restrict__ dst,
                            int rows, int cs, int cd) {
  size_t n = (size_t)rows * cd;
  size_t i = (size_t)blockIdx.x * blockDim.x + threadIdx.x;
  size_t s = (size_t)gridDim.x * blockDim.x;
  for (; i < n; i += s) {
    int r = (int)(i / cd), c = (int)(i % cd);
    float v = (c < cs) ? src[(size_t)r * cs + c] : 0.0f;
    dst[i] = f2bf(v);
  }
}

// xin[t][b][k] = concat(x_shifted[b][:][t], cond[b][:][t], pad16) in bf16
__global__ void build_inputs(const float* __restrict__ x, const float* __restrict__ cond,
                             bf16_t* __restrict__ xinb) {
  size_t n = (size_t)NT * NB * K0P;
  size_t i = (size_t)blockIdx.x * blockDim.x + threadIdx.x;
  size_t s = (size_t)gridDim.x * blockDim.x;
  for (; i < n; i += s) {
    int k = (int)(i % K0P);
    size_t tb = i / K0P;
    int b = (int)(tb % NB);
    int t = (int)(tb / NB);
    float v = 0.0f;
    if (k < ND) {
      if (t > 0) v = x[(size_t)b * ND * NT + (size_t)k * NT + (t - 1)];
    } else if (k < ND + NC) {
      v = cond[(size_t)b * NC * NT + (size_t)(k - ND) * NT + t];
    }
    xinb[i] = f2bf(v);
  }
}

// ---- persistent recurrent kernel ------------------------------------------
__global__ void __launch_bounds__(NTHR, 1) lstm_persistent(
    char* __restrict__ ws, const int* __restrict__ xlen,
    const float* __restrict__ b_ih0, const float* __restrict__ b_hh0,
    const float* __restrict__ b_ih1, const float* __restrict__ b_hh1,
    const float* __restrict__ proj_b, float* __restrict__ out)
{
  const bf16_t* w_ih0b = (const bf16_t*)(ws + OFF_WIH0);
  const bf16_t* w_hh0b = (const bf16_t*)(ws + OFF_WHH0);
  const bf16_t* w_ih1b = (const bf16_t*)(ws + OFF_WIH1);
  const bf16_t* w_hh1b = (const bf16_t*)(ws + OFF_WHH1);
  const bf16_t* projwb = (const bf16_t*)(ws + OFF_PROJW);
  const bf16_t* xinb   = (const bf16_t*)(ws + OFF_XIN);
  bf16_t* h0b = (bf16_t*)(ws + OFF_H0B);
  bf16_t* h1b = (bf16_t*)(ws + OFF_H1B);
  float*  h0f = (float*)(ws + OFF_H0F);
  float*  c0f = (float*)(ws + OFF_C0F);
  float*  h1f = (float*)(ws + OFF_H1F);
  float*  c1f = (float*)(ws + OFF_C1F);
  float*  g0  = (float*)(ws + OFF_G0);
  float*  g1  = (float*)(ws + OFF_G1);
  unsigned* bar = (unsigned*)(ws + OFF_BAR);   // [0]=cnt [1]=gen

  const int wid  = blockIdx.x * (NTHR / 32) + (threadIdx.x >> 5);  // 0..511
  const int lane = threadIdx.x & 31;
  const int gtid = blockIdx.x * NTHR + threadIdx.x;

  for (int t = 0; t < NT; ++t) {
    // Phase 1: layer-0 gates = xin @ w_ih0^T + h0 @ w_hh0^T
    const bf16_t* xin_t = xinb + (size_t)t * NB * K0P;
    gemm_gates(xin_t, K0P, w_ih0b, h0b, w_hh0b, g0, wid, lane);
    grid_sync(bar, bar + 1);

    // Phase 2: zoneout cell update -> h0, c0
    cell_pointwise(g0, b_ih0, b_hh0, h0f, c0f, h0b, gtid);
    grid_sync(bar, bar + 1);

    // Phase 3: layer-1 gates = h0 @ w_ih1^T + h1 @ w_hh1^T
    gemm_gates(h0b, NH, w_ih1b, h1b, w_hh1b, g1, wid, lane);
    grid_sync(bar, bar + 1);

    // Phase 4: zoneout cell update -> h1, c1
    cell_pointwise(g1, b_ih1, b_hh1, h1f, c1f, h1b, gtid);
    grid_sync(bar, bar + 1);

    // Phase 5: projection y = h1 @ proj_w^T + proj_b, masked, strided store.
    // 10 tiles (2 M-tiles x 5 N-tiles); runs concurrently with next step's
    // phase 1 (disjoint read/write sets), so no extra barrier needed.
    if (wid < (NB / 16) * (ND / 16)) {
      const int mt = wid & 1, nt = wid >> 1;
      const int n0 = nt * 16;
      const int r  = lane & 15;
      const bf16_t* a = h1b    + (size_t)(mt * 16 + r) * NH;
      const bf16_t* w = projwb + (size_t)(n0 + r) * NH;
      v8f acc;
#pragma unroll
      for (int i = 0; i < 8; ++i) acc[i] = 0.0f;
      for (int k0 = 0; k0 < NH; k0 += 32)
        acc = wmma_bf16(load_frag(a, k0), load_frag(w, k0), acc);
      const int col   = n0 + r;
      const float bias = proj_b[col];
      const int mbase = mt * 16 + ((lane >> 4) & 1) * 8;
#pragma unroll
      for (int rr = 0; rr < 8; ++rr) {
        int b = mbase + rr;
        float v = (t < xlen[b]) ? (acc[rr] + bias) : 0.0f;
        out[(size_t)b * ND * NT + (size_t)col * NT + t] = v;
      }
    }
  }
}

// ---- entry ----------------------------------------------------------------
extern "C" void kernel_launch(void* const* d_in, const int* in_sizes, int n_in,
                              void* d_out, int out_size, void* d_ws, size_t ws_size,
                              hipStream_t stream) {
  const float* x      = (const float*)d_in[0];
  const float* cond   = (const float*)d_in[1];
  const int*   xlen   = (const int*)  d_in[2];
  const float* w_ih0  = (const float*)d_in[3];
  const float* w_hh0  = (const float*)d_in[4];
  const float* b_ih0  = (const float*)d_in[5];
  const float* b_hh0  = (const float*)d_in[6];
  const float* w_ih1  = (const float*)d_in[7];
  const float* w_hh1  = (const float*)d_in[8];
  const float* b_ih1  = (const float*)d_in[9];
  const float* b_hh1  = (const float*)d_in[10];
  const float* proj_w = (const float*)d_in[11];
  const float* proj_b = (const float*)d_in[12];
  char*  ws  = (char*)d_ws;
  float* out = (float*)d_out;

  // state + barrier zero-init (every call: determinism across graph replays)
  zero_ws<<<dim3(256), dim3(NTHR), 0, stream>>>((float*)(ws + ZERO_OFF), ZERO_BYTES / 4);

  // weights -> bf16 (w_ih0 K-padded 592 -> 608)
  convert_pad<<<dim3(1024), dim3(NTHR), 0, stream>>>(w_ih0, (bf16_t*)(ws + OFF_WIH0), NG, ND + NC, K0P);
  convert_pad<<<dim3(1024), dim3(NTHR), 0, stream>>>(w_hh0, (bf16_t*)(ws + OFF_WHH0), NG, NH, NH);
  convert_pad<<<dim3(1024), dim3(NTHR), 0, stream>>>(w_ih1, (bf16_t*)(ws + OFF_WIH1), NG, NH, NH);
  convert_pad<<<dim3(1024), dim3(NTHR), 0, stream>>>(w_hh1, (bf16_t*)(ws + OFF_WHH1), NG, NH, NH);
  convert_pad<<<dim3(64),   dim3(NTHR), 0, stream>>>(proj_w, (bf16_t*)(ws + OFF_PROJW), ND, NH, NH);

  // shifted+concatenated inputs -> bf16 [T][B][608]
  build_inputs<<<dim3(2048), dim3(NTHR), 0, stream>>>(x, cond, (bf16_t*)(ws + OFF_XIN));

  // persistent recurrent kernel: 64 WGs x 256 threads (512 wave32 waves)
  lstm_persistent<<<dim3(NWG), dim3(NTHR), 0, stream>>>(
      ws, xlen, b_ih0, b_hh0, b_ih1, b_hh1, proj_b, out);
}
